// GCN_node_45801531245068
// MI455X (gfx1250) — matrix operand order repaired and checked by reading, not compiled, not run on hardware
//
#include <hip/hip_runtime.h>
#include <hip/hip_bf16.h>

typedef _Float16 v16h __attribute__((ext_vector_type(16)));
typedef float    v8f  __attribute__((ext_vector_type(8)));

#define BN_EPS 1e-5f

// ---------------------------------------------------------------------------
// Utility: zero a float buffer
// ---------------------------------------------------------------------------
__global__ void zero_f32(float* __restrict__ p, int n) {
    int t = blockIdx.x * blockDim.x + threadIdx.x;
    if (t < n) p[t] = 0.0f;
}

// ---------------------------------------------------------------------------
// In-degree (self-loops folded in later as +1)
// ---------------------------------------------------------------------------
__global__ void degree_kernel(const int* __restrict__ dst, float* __restrict__ deg, int E) {
    int t = blockIdx.x * blockDim.x + threadIdx.x;
    if (t < E) atomicAdd(&deg[dst[t]], 1.0f);
}

// dis[i] = rsqrt(deg[i] + 1)   (in-place over deg buffer)
__global__ void dis_kernel(float* __restrict__ deg, int N) {
    int t = blockIdx.x * blockDim.x + threadIdx.x;
    if (t < N) deg[t] = rsqrtf(deg[t] + 1.0f);
}

// ---------------------------------------------------------------------------
// WMMA GEMM:  H[N x 64] = X[N x 64] @ W[64 x 64]  (f16 inputs, f32 accumulate)
// 4 waves / block; wave w computes the 16x16 tile (rows blockIdx*16, cols w*16).
// K = 64 -> two v_wmma_f32_16x16x32_f16 steps (k0 = 0, 32).
//
// Layouts (wave32, g = lane/16):
//  A 16x32: lane -> row m = lane%16. v16h element j: VGPR v=j>>1, odd=j&1;
//           K = j + 16g (same contiguous order as B)
//           => one contiguous K chunk [16g, 16g+16) for j=0..15.
//  B 32x16: lane -> col n = lane%16. v16h element j: K = j + 16g.
//  C/D 16x16 f32: VGPR r -> row r + 8g, col n = lane%16.
// ---------------------------------------------------------------------------
__global__ __launch_bounds__(128)
void gemm_wmma(const float* __restrict__ X, const float* __restrict__ W,
               float* __restrict__ H, int N) {
    const int wave = threadIdx.x >> 5;
    const int lane = threadIdx.x & 31;
    const int m0 = blockIdx.x * 16;
    if (m0 >= N) return;
    const int n0 = wave * 16;
    const int ml = lane & 15;   // row (A) / col (B/D)
    const int g  = lane >> 4;   // lane half

    int arow = m0 + ml;
    if (arow >= N) arow = N - 1;  // clamp: pollutes only unstored C rows
    const float* __restrict__ xrow = X + (size_t)arow * 64;

    v8f c = {};
#pragma unroll
    for (int k0 = 0; k0 < 64; k0 += 32) {
        v16h a, b;
        const float* xk = xrow + k0;
#pragma unroll
        for (int j = 0; j < 8; ++j) {
            a[j]     = (_Float16)xk[16 * g + j];       // K = 16g..16g+7
            a[8 + j] = (_Float16)xk[16 * g + 8 + j];   // K = 16g+8..16g+15
        }
#pragma unroll
        for (int j = 0; j < 16; ++j) {
            b[j] = (_Float16)W[(k0 + 16 * g + j) * 64 + n0 + ml];  // K = j+16g
        }
        // 8 args: (neg_a, A, neg_b, B, c_mod, C, reuse_a, reuse_b)
        c = __builtin_amdgcn_wmma_f32_16x16x32_f16(false, a, false, b,
                                                   (short)0, c, false, false);
    }

    // Epilogue: fast unguarded path for full tiles (all but possibly the last
    // M-tile), guarded path otherwise. Avoids 8x v_cmpx/EXEC-juggling stores.
    const int mbase = m0 + (g << 3);
    float* __restrict__ hout = H + (size_t)mbase * 64 + n0 + ml;
    if (mbase + 7 < N) {
#pragma unroll
        for (int r = 0; r < 8; ++r) {
            hout[(size_t)r * 64] = c[r];
        }
    } else {
#pragma unroll
        for (int r = 0; r < 8; ++r) {
            if (mbase + r < N) hout[(size_t)r * 64] = c[r];
        }
    }
}

// ---------------------------------------------------------------------------
// Edge scatter: OUT[dst] += H[src] * dis[src]*dis[dst]
// One thread per (edge, 4-feature chunk): 16 threads/edge, float4 gather.
// ---------------------------------------------------------------------------
__global__ __launch_bounds__(256)
void scatter_edges(const float* __restrict__ H, const int* __restrict__ src,
                   const int* __restrict__ dst, const float* __restrict__ dis,
                   float* __restrict__ OUT, int E) {
    const int t = blockIdx.x * blockDim.x + threadIdx.x;
    if (t >= E * 16) return;
    const int e = t >> 4;
    const int f = (t & 15) * 4;
    const int s = src[e];
    const int d = dst[e];
    const float nrm = dis[s] * dis[d];
    const float4 hv = *reinterpret_cast<const float4*>(H + (size_t)s * 64 + f);
    float* o = OUT + (size_t)d * 64 + f;
    atomicAdd(o + 0, hv.x * nrm);
    atomicAdd(o + 1, hv.y * nrm);
    atomicAdd(o + 2, hv.z * nrm);
    atomicAdd(o + 3, hv.w * nrm);
}

// ---------------------------------------------------------------------------
// Self-loop contribution + bias: OUT[i][f] += H[i][f]*dis[i]^2 + b[f]
// ---------------------------------------------------------------------------
__global__ __launch_bounds__(256)
void selfloop_bias(const float* __restrict__ H, const float* __restrict__ dis,
                   const float* __restrict__ bias, float* __restrict__ OUT, int N) {
    const int t = blockIdx.x * blockDim.x + threadIdx.x;
    if (t >= N * 64) return;
    const int i = t >> 6;
    const int f = t & 63;
    const float di = dis[i];
    OUT[t] += H[t] * di * di + bias[f];
}

// ---------------------------------------------------------------------------
// BatchNorm statistics: sums[f] = sum_i X[i][f], sums[64+f] = sum_i X[i][f]^2
// Register accumulate -> LDS combine -> 128 atomics per block.
// ---------------------------------------------------------------------------
__global__ __launch_bounds__(256)
void bn_stats(const float* __restrict__ X, float* __restrict__ sums, int N) {
    __shared__ float ssum[256];
    __shared__ float ssq[256];
    const int f = threadIdx.x & 63;
    const int rg = threadIdx.x >> 6;  // 0..3
    float s = 0.0f, q = 0.0f;
    for (int r = blockIdx.x * 4 + rg; r < N; r += gridDim.x * 4) {
        const float v = X[(size_t)r * 64 + f];
        s += v;
        q += v * v;
    }
    ssum[threadIdx.x] = s;
    ssq[threadIdx.x] = q;
    __syncthreads();
    if (rg == 0) {
        s = ssum[f] + ssum[f + 64] + ssum[f + 128] + ssum[f + 192];
        q = ssq[f] + ssq[f + 64] + ssq[f + 128] + ssq[f + 192];
        atomicAdd(&sums[f], s);
        atomicAdd(&sums[64 + f], q);
    }
}

// ---------------------------------------------------------------------------
// BatchNorm (biased var, training batch stats) + ReLU
// ---------------------------------------------------------------------------
__global__ __launch_bounds__(256)
void bn_relu(const float* __restrict__ X, const float* __restrict__ sums,
             const float* __restrict__ gamma, const float* __restrict__ beta,
             float* __restrict__ Y, int N) {
    const int t = blockIdx.x * blockDim.x + threadIdx.x;
    if (t >= N * 64) return;
    const int f = t & 63;
    const float invN = 1.0f / (float)N;
    const float mu = sums[f] * invN;
    const float var = fmaxf(sums[64 + f] * invN - mu * mu, 0.0f);
    const float y = (X[t] - mu) * rsqrtf(var + BN_EPS) * gamma[f] + beta[f];
    Y[t] = fmaxf(y, 0.0f);
}

// ---------------------------------------------------------------------------
// Launcher
// ---------------------------------------------------------------------------
extern "C" void kernel_launch(void* const* d_in, const int* in_sizes, int n_in,
                              void* d_out, int out_size, void* d_ws, size_t ws_size,
                              hipStream_t stream) {
    const float* x   = (const float*)d_in[0];
    const int*   ei  = (const int*)d_in[1];
    const float* W1  = (const float*)d_in[2];
    const float* b1  = (const float*)d_in[3];
    const float* g1  = (const float*)d_in[4];
    const float* be1 = (const float*)d_in[5];
    const float* W2  = (const float*)d_in[6];
    const float* b2  = (const float*)d_in[7];
    const float* g2  = (const float*)d_in[8];
    const float* be2 = (const float*)d_in[9];
    float* out = (float*)d_out;

    const int N = in_sizes[0] / 64;
    const int E = in_sizes[1] / 2;
    const int* src = ei;
    const int* dst = ei + E;

    // Workspace layout (floats): dis | bufH | bufG | bufAgg | sums(128)
    float* ws     = (float*)d_ws;
    float* dis    = ws;
    float* bufH   = dis + N;
    float* bufG   = bufH + (size_t)N * 64;
    float* bufAgg = bufG + (size_t)N * 64;
    float* sums   = bufAgg + (size_t)N * 64;  // 128 floats (sum | sumsq)

    const int NF = N * 64;
    const int gNF  = (NF + 255) / 256;
    const int gN   = (N + 255) / 256;
    const int gE   = (E + 255) / 256;
    const int gE16 = (E * 16 + 255) / 256;  // E*16 = 25.6M, fits int
    const int gZ   = (NF + 128 + 255) / 256;
    const int gM   = (N + 15) / 16;

    // --- degree / normalization ---
    zero_f32<<<gN, 256, 0, stream>>>(dis, N);
    degree_kernel<<<gE, 256, 0, stream>>>(dst, dis, E);
    dis_kernel<<<gN, 256, 0, stream>>>(dis, N);

    // ============================ Layer 1 ============================
    gemm_wmma<<<gM, 128, 0, stream>>>(x, W1, bufH, N);
    zero_f32<<<gZ, 256, 0, stream>>>(bufAgg, NF + 128);
    scatter_edges<<<gE16, 256, 0, stream>>>(bufH, src, dst, dis, bufAgg, E);
    selfloop_bias<<<gNF, 256, 0, stream>>>(bufH, dis, b1, bufAgg, N);
    bn_stats<<<256, 256, 0, stream>>>(bufAgg, sums, N);
    bn_relu<<<gNF, 256, 0, stream>>>(bufAgg, sums, g1, be1, bufH, N);

    // ============================ Layer 2 ============================
    gemm_wmma<<<gM, 128, 0, stream>>>(bufH, W2, bufG, N);
    zero_f32<<<gZ, 256, 0, stream>>>(bufAgg, NF + 128);
    scatter_edges<<<gE16, 256, 0, stream>>>(bufG, src, dst, dis, bufAgg, E);
    selfloop_bias<<<gNF, 256, 0, stream>>>(bufG, dis, b2, bufAgg, N);
    bn_stats<<<256, 256, 0, stream>>>(bufAgg, sums, N);
    bn_relu<<<gNF, 256, 0, stream>>>(bufAgg, sums, g2, be2, out, N);
}
